// InputFeedRNNDecoder_43774306681475
// MI455X (gfx1250) — matrix operand order repaired and checked by reading, not compiled, not run on hardware
//
#include <hip/hip_runtime.h>
#include <math.h>

typedef __bf16 bf16;
typedef __attribute__((ext_vector_type(16))) __bf16 v16bf;
typedef __attribute__((ext_vector_type(8)))  __bf16 v8bf;
typedef __attribute__((ext_vector_type(8)))  float  v8f;

#define B_ 64
#define T_ 64
#define S_ 128
#define H_ 1024
#define E_ 1024

// ---------------- prep kernels ----------------

// out[n, 0:Ka] = bf16(Wa[n, :]); out[n, Ka:Ka+Kb] = bf16(Wb[n, :])
// (pass Kb==0 for a plain fp32->bf16 convert)
__global__ void k_pack_cat(const float* __restrict__ Wa, int Ka,
                           const float* __restrict__ Wb, int Kb,
                           bf16* __restrict__ out, long total) {
  long idx = (long)blockIdx.x * blockDim.x + threadIdx.x;
  if (idx >= total) return;
  int K = Ka + Kb;
  long n = idx / K;
  int  k = (int)(idx % K);
  float v = (k < Ka) ? Wa[n * (long)Ka + k] : Wb[n * (long)Kb + (k - Ka)];
  out[idx] = (bf16)v;
}

// emb_all[b,t,e] = bf16(table[tgt[b,t], e])
__global__ void k_emb_gather(const int* __restrict__ tgt,
                             const float* __restrict__ table,
                             bf16* __restrict__ out) {
  long idx = (long)blockIdx.x * blockDim.x + threadIdx.x;
  long total = (long)B_ * T_ * E_;
  if (idx >= total) return;
  int e  = (int)(idx % E_);
  long bt = idx / E_;
  int tok = tgt[bt];
  out[idx] = (bf16)table[(long)tok * E_ + e];
}

// initialize recurrent state buffers
__global__ void k_init(const float* __restrict__ h0, const float* __restrict__ c0,
                       bf16* __restrict__ x0, bf16* __restrict__ x1,
                       float* __restrict__ c0s, float* __restrict__ c1s) {
  int idx = blockIdx.x * blockDim.x + threadIdx.x;
  if (idx >= B_ * H_) return;
  int b = idx / H_, h = idx % H_;
  x0[(long)b * 3072 + 1024 + h] = (bf16)0.0f;                 // input feed = 0
  x0[(long)b * 3072 + 2048 + h] = (bf16)h0[idx];              // h layer0
  x1[(long)b * 2048 + 1024 + h] = (bf16)h0[B_ * H_ + idx];    // h layer1
  c0s[idx] = c0[idx];
  c1s[idx] = c0[B_ * H_ + idx];
}

// per-step: x0[:, 0:1024] = emb_all[:, t, :]
__global__ void k_set_emb(const bf16* __restrict__ emb_all, bf16* __restrict__ x0, int t) {
  int idx = blockIdx.x * blockDim.x + threadIdx.x;
  if (idx >= B_ * E_) return;
  int b = idx / E_, e = idx % E_;
  x0[(long)b * 3072 + e] = emb_all[((long)b * T_ + t) * E_ + e];
}

// ---------------- WMMA GEMM: C(64 x N) = A(64 x K, lda) * W(N x K)^T + bias ----------------
// Each wave computes one 16x32 output block: 1 M-tile x NT=2 N-tiles.
//  - A fragment loaded once per k-step, shared by both WMMAs (halves A re-read traffic)
//  - two independent accumulators give the WMMA pipe issue slack
//  - weight stream prefetched at a CONSTANT +512B offset so the prefetch folds into
//    the load base registers' immediate offset (no per-iteration address arithmetic);
//    prefetch is speculative on gfx1250 and the tail overrun stays inside d_ws.
#define NT 2
__global__ void k_wmma_gemm(const bf16* __restrict__ A, int lda,
                            const bf16* __restrict__ W,
                            const float* __restrict__ bias,
                            float* __restrict__ C, int N, int K) {
  const int lane = threadIdx.x & 31;
  const int wave = threadIdx.x >> 5;
  const int wid  = blockIdx.x * (blockDim.x >> 5) + wave;
  const int mtile = wid & 3;               // M = 64 -> 4 tiles
  const int ngrp  = wid >> 2;              // group of NT n-tiles (32 cols)
  if (ngrp >= (N >> 5)) return;

  const int hi = lane >> 4;                // lane half: 0 or 1
  const int lm = lane & 15;
  // A fragment: lane holds row M=lm; halves [0..7] = K=kA..kA+7, [8..15] = K=kA+16..kA+23
  const bf16* Ap = A + (long)(mtile * 16 + lm) * lda + hi * 8;
  // B fragments: lane holds column N=lm of each tile; same K layout as A
  const bf16* Wp[NT];
#pragma unroll
  for (int j = 0; j < NT; ++j)
    Wp[j] = W + (long)(ngrp * (16 * NT) + j * 16 + lm) * K + hi * 8;

  v8f acc[NT] = {};
  for (int k0 = 0; k0 < K; k0 += 32) {
    // prefetch the weight stream 512B ahead at a constant offset
#pragma unroll
    for (int j = 0; j < NT; ++j)
      __builtin_prefetch((const void*)(Wp[j] + k0 + 256), 0, 1);

    v8bf a0 = *(const v8bf*)(Ap + k0);
    v8bf a1 = *(const v8bf*)(Ap + k0 + 16);
    v16bf a;
#pragma unroll
    for (int i = 0; i < 8; ++i) { a[i] = a0[i]; a[8 + i] = a1[i]; }

#pragma unroll
    for (int j = 0; j < NT; ++j) {
      v8bf b0 = *(const v8bf*)(Wp[j] + k0);
      v8bf b1 = *(const v8bf*)(Wp[j] + k0 + 16);
      v16bf bm;
#pragma unroll
      for (int i = 0; i < 8; ++i) { bm[i] = b0[i]; bm[8 + i] = b1[i]; }
      acc[j] = __builtin_amdgcn_wmma_f32_16x16x32_bf16(
          false, a, false, bm, (short)0, acc[j], false, false);
    }
  }

#pragma unroll
  for (int j = 0; j < NT; ++j) {
    int col = ngrp * (16 * NT) + j * 16 + lm;
    float bv = bias ? bias[col] : 0.0f;
#pragma unroll
    for (int r = 0; r < 8; ++r) {
      int row = mtile * 16 + hi * 8 + r;   // C layout: VGPR r -> M = r (+8 for hi lanes)
      C[(long)row * N + col] = acc[j][r] + bv;
    }
  }
}

// ---------------- LSTM elementwise ----------------
// gate order i,f,g,o.  c updated in place (fp32); h written as bf16 to up to 2 slots.
__global__ void k_lstm(const float* __restrict__ g, float* __restrict__ c,
                       bf16* __restrict__ h_dst0, int stride0, int off0,
                       bf16* __restrict__ h_dst1, int stride1, int off1) {
  int idx = blockIdx.x * blockDim.x + threadIdx.x;
  if (idx >= B_ * H_) return;
  int b = idx / H_, h = idx % H_;
  const float* gb = g + (long)b * 4 * H_;
  float i_ = gb[h], f_ = gb[H_ + h], g_ = gb[2 * H_ + h], o_ = gb[3 * H_ + h];
  float si = 1.0f / (1.0f + __expf(-i_));
  float sf = 1.0f / (1.0f + __expf(-f_));
  float so = 1.0f / (1.0f + __expf(-o_));
  float cn = sf * c[idx] + si * tanhf(g_);
  c[idx] = cn;
  bf16 hb = (bf16)(so * tanhf(cn));
  h_dst0[(long)b * stride0 + off0 + h] = hb;
  if (h_dst1) h_dst1[(long)b * stride1 + off1 + h] = hb;
}

// ---------------- attention: scores -> masked softmax -> ctx ----------------
// one block (128 threads = 4 waves) per batch element.
__global__ void k_attn(const float* __restrict__ q,      // (B,H)
                       const float* __restrict__ enc,    // (B,S,H)
                       const int* __restrict__ src_len,
                       float* __restrict__ attn_std,     // (T,B,S)
                       bf16* __restrict__ xo,            // (B,2048): ctx -> cols 0..1023
                       int t) {
  const int b = blockIdx.x;
  const int tid = threadIdx.x;
  const int lane = tid & 31, wave = tid >> 5;
  __shared__ float s_sc[S_];
  __shared__ float s_red[4];

  const float* qb = q + (long)b * H_;
  const float* eb = enc + (long)b * S_ * H_;
  const int len = src_len[b];

  // scores[s] = q . enc[b,s,:]
  for (int s = wave; s < S_; s += 4) {
    const float* es = eb + (long)s * H_;
    float p = 0.0f;
    for (int k = lane; k < H_; k += 32) p += qb[k] * es[k];
#pragma unroll
    for (int off = 16; off; off >>= 1) p += __shfl_xor(p, off, 32);
    if (lane == 0) s_sc[s] = (s < len) ? p : -1e9f;
  }
  __syncthreads();

  // masked softmax over S=128 (thread tid owns s=tid)
  float v = s_sc[tid];
  float m = v;
#pragma unroll
  for (int off = 16; off; off >>= 1) m = fmaxf(m, __shfl_xor(m, off, 32));
  if (lane == 0) s_red[wave] = m;
  __syncthreads();
  m = fmaxf(fmaxf(s_red[0], s_red[1]), fmaxf(s_red[2], s_red[3]));
  float e = __expf(v - m);
  float sum = e;
#pragma unroll
  for (int off = 16; off; off >>= 1) sum += __shfl_xor(sum, off, 32);
  __syncthreads();
  if (lane == 0) s_red[wave] = sum;
  __syncthreads();
  sum = s_red[0] + s_red[1] + s_red[2] + s_red[3];
  float al = e / sum;
  s_sc[tid] = al;
  attn_std[((long)t * B_ + b) * S_ + tid] = al;
  __syncthreads();

  // ctx[h] = sum_s align[s] * enc[b,s,h]
  for (int h = tid; h < H_; h += 128) {
    float cx = 0.0f;
    for (int s = 0; s < S_; ++s) cx += s_sc[s] * eb[(long)s * H_ + h];
    xo[(long)b * 2048 + h] = (bf16)cx;
  }
}

// ---------------- output tanh + feedback ----------------
__global__ void k_out(const float* __restrict__ ao, float* __restrict__ dec_outs,
                      bf16* __restrict__ x0, int t) {
  int idx = blockIdx.x * blockDim.x + threadIdx.x;
  if (idx >= B_ * H_) return;
  int b = idx / H_, h = idx % H_;
  float v = tanhf(ao[idx]);
  dec_outs[((long)b * T_ + t) * H_ + h] = v;            // (B,T,H)
  x0[(long)b * 3072 + 1024 + h] = (bf16)v;              // input feed for t+1
}

// ---------------- host ----------------
extern "C" void kernel_launch(void* const* d_in, const int* in_sizes, int n_in,
                              void* d_out, int out_size, void* d_ws, size_t ws_size,
                              hipStream_t stream) {
  const int*   tgt     = (const int*)  d_in[0];
  const int*   src_len = (const int*)  d_in[1];
  const float* enc     = (const float*)d_in[2];
  const float* h0      = (const float*)d_in[3];
  const float* c0      = (const float*)d_in[4];
  const float* embt    = (const float*)d_in[5];
  const float* W_ih0   = (const float*)d_in[6];
  const float* W_hh0   = (const float*)d_in[7];
  const float* b0      = (const float*)d_in[8];
  const float* W_ih1   = (const float*)d_in[9];
  const float* W_hh1   = (const float*)d_in[10];
  const float* b1      = (const float*)d_in[11];
  const float* W_a     = (const float*)d_in[12];
  const float* W_out   = (const float*)d_in[13];

  float* dec_outs = (float*)d_out;                       // (B,T,H)
  float* attn_std = (float*)d_out + (long)B_ * T_ * H_;  // (T,B,S)

  // workspace carve-up (bf16 weight cache + activations), 256B aligned
  char* p = (char*)d_ws;
  auto take = [&](size_t bytes) { char* r = p; p += (bytes + 255) & ~(size_t)255; return r; };
  bf16* Wcat0   = (bf16*)take((size_t)4096 * 3072 * 2);  // [W_ih0 | W_hh0]
  bf16* Wcat1   = (bf16*)take((size_t)4096 * 2048 * 2);  // [W_ih1 | W_hh1]
  bf16* Wa_bf   = (bf16*)take((size_t)1024 * 1024 * 2);
  bf16* Wout_bf = (bf16*)take((size_t)1024 * 2048 * 2);
  bf16* emb_all = (bf16*)take((size_t)B_ * T_ * E_ * 2);
  bf16* x0      = (bf16*)take((size_t)B_ * 3072 * 2);    // [emb | feed | h0]
  bf16* x1      = (bf16*)take((size_t)B_ * 2048 * 2);    // [h0n | h1]
  bf16* xo      = (bf16*)take((size_t)B_ * 2048 * 2);    // [ctx | h1n]
  float* g0     = (float*)take((size_t)B_ * 4096 * 4);
  float* g1     = (float*)take((size_t)B_ * 4096 * 4);
  float* qbuf   = (float*)take((size_t)B_ * H_ * 4);
  float* ao     = (float*)take((size_t)B_ * H_ * 4);
  float* c0s    = (float*)take((size_t)B_ * H_ * 4);
  float* c1s    = (float*)take((size_t)B_ * H_ * 4);

  const int TB = 256;
  auto blocks = [](long total, int tb) { return (unsigned)((total + tb - 1) / tb); };

  // --- one-time prep (per call; deterministic) ---
  {
    long t0 = (long)4096 * 3072;
    k_pack_cat<<<blocks(t0, TB), TB, 0, stream>>>(W_ih0, 2048, W_hh0, 1024, Wcat0, t0);
    long t1 = (long)4096 * 2048;
    k_pack_cat<<<blocks(t1, TB), TB, 0, stream>>>(W_ih1, 1024, W_hh1, 1024, Wcat1, t1);
    long t2 = (long)1024 * 1024;
    k_pack_cat<<<blocks(t2, TB), TB, 0, stream>>>(W_a, 1024, W_a, 0, Wa_bf, t2);
    long t3 = (long)1024 * 2048;
    k_pack_cat<<<blocks(t3, TB), TB, 0, stream>>>(W_out, 2048, W_out, 0, Wout_bf, t3);
    long t4 = (long)B_ * T_ * E_;
    k_emb_gather<<<blocks(t4, TB), TB, 0, stream>>>(tgt, embt, emb_all);
    k_init<<<blocks(B_ * H_, TB), TB, 0, stream>>>(h0, c0, x0, x1, c0s, c1s);
  }

  // --- sequential decode loop ---
  // wave count per GEMM: waves = 4 * (N/32); gates: 512 waves -> 64 blocks;
  // N=1024 GEMMs: 128 waves -> 16 blocks (8 waves / 256 threads per block)
  for (int t = 0; t < T_; ++t) {
    k_set_emb<<<blocks(B_ * E_, TB), TB, 0, stream>>>(emb_all, x0, t);

    // LSTM layer 0: g0 = x0 @ Wcat0^T + b0   (N=4096, K=3072)
    k_wmma_gemm<<<64, 256, 0, stream>>>(x0, 3072, Wcat0, b0, g0, 4096, 3072);
    k_lstm<<<blocks(B_ * H_, TB), TB, 0, stream>>>(g0, c0s, x0, 3072, 2048, x1, 2048, 0);

    // LSTM layer 1: g1 = x1 @ Wcat1^T + b1   (N=4096, K=2048)
    k_wmma_gemm<<<64, 256, 0, stream>>>(x1, 2048, Wcat1, b1, g1, 4096, 2048);
    k_lstm<<<blocks(B_ * H_, TB), TB, 0, stream>>>(g1, c1s, x1, 2048, 1024, xo, 2048, 1024);

    // q = h1n @ W_a^T  (A = x1[:,1024:2048], lda=2048; N=K=1024)
    k_wmma_gemm<<<16, 256, 0, stream>>>(x1 + 1024, 2048, Wa_bf, nullptr, qbuf, 1024, 1024);

    // attention (scores, masked softmax, ctx)
    k_attn<<<B_, 128, 0, stream>>>(qbuf, enc, src_len, attn_std, xo, t);

    // attn_h = tanh([ctx | h1n] @ W_out^T)   (N=1024, K=2048)
    k_wmma_gemm<<<16, 256, 0, stream>>>(xo, 2048, Wout_bf, nullptr, ao, 1024, 2048);
    k_out<<<blocks(B_ * H_, TB), TB, 0, stream>>>(ao, dec_outs, x0, t);
  }
}
